// C3D_15470472200649
// MI455X (gfx1250) — matrix operand order repaired and checked by reading, not compiled, not run on hardware
//
#include <hip/hip_runtime.h>
#include <hip/hip_bf16.h>
#include <cmath>

typedef __attribute__((ext_vector_type(16))) _Float16 v16h;
typedef __attribute__((ext_vector_type(8)))  _Float16 v8h;
typedef __attribute__((ext_vector_type(4)))  _Float16 v4h;
typedef __attribute__((ext_vector_type(8)))  float    v8f;
typedef __attribute__((ext_vector_type(4)))  float    v4f;

// ---------------------------------------------------------------------------
// Fragment loaders (CDNA5 ISA 7.12.2, wave32)
// A 16x32 f16: lanes 0-15 row M=lane, K={0..7,16..23}; lanes 16-31 row
// M=lane-16, K={8..15,24..31}. Caller pre-biases pointer +8 for lanes>=16.
__device__ __forceinline__ v16h load_fragA(const _Float16* p) {
  v8h lo = *(const v8h*)(p);
  v8h hi = *(const v8h*)(p + 16);
  v16h r;
#pragma unroll
  for (int i = 0; i < 8; ++i) { r[i] = lo[i]; r[i + 8] = hi[i]; }
  return r;
}
// B 32x16 f16: lane holds column N=lane%16, K={0..7,16..23} lanes 0-15,
// K={8..15,24..31} lanes 16-31. Caller pre-biases pointer +8 for lanes>=16.
__device__ __forceinline__ v16h load_fragB(const _Float16* p) {
  v8h lo = *(const v8h*)(p);
  v8h hi = *(const v8h*)(p + 16);
  v16h r;
#pragma unroll
  for (int i = 0; i < 8; ++i) { r[i] = lo[i]; r[i + 8] = hi[i]; }
  return r;
}

// ---------------------------------------------------------------------------
// Implicit-GEMM 3x3x3 conv (pad 1, stride 1) via WMMA, NDHWC activations.
//   x : [D][H][W][ICp] f16 (one clip)    w : [OC][Kpad] f16, K = tap*ICp+ic
//   y : [D][H][W][OC]  f16               y = relu(scale*(conv+bias))
// grid: (ceil(P/32), OC/(NWAVES*16)), block NWAVES*32.
// VEC path (ICp>=8): per K-step each thread gathers ELEM contiguous channels
// of one spatial tap with a single vector load (tap = kk>>icShift).
template <int NWAVES, bool VEC>
__global__ void __launch_bounds__(NWAVES * 32)
conv3d_wmma(const _Float16* __restrict__ x, const _Float16* __restrict__ w,
            const float* __restrict__ bias, _Float16* __restrict__ y,
            int ICp, int icShift, int ICreal, int D, int H, int W,
            int Kpad, int OC, float scale) {
  constexpr int NT = NWAVES * 32;
  constexpr int ELEM = 1024 / NT;            // halves gathered per thread/step
  const int HW = H * W;
  const int P  = D * HW;
  const int p0 = blockIdx.x * 32;
  const int ocBase = blockIdx.y * (NWAVES * 16);

  __shared__ __align__(16) _Float16 Bs[32][40];   // [N][K], row pad = 80B

  const int tid  = threadIdx.x;
  const int lane = tid & 31;
  const int wv   = tid >> 5;

  // B-fill coordinates: ELEM consecutive K of one position
  const int fIdx  = tid * ELEM;
  const int nFill = fIdx >> 5;
  const int kCol  = fIdx & 31;
  const int pFill = p0 + nFill;
  const bool pValid = (pFill < P);
  int fd = 0, fh = 0, fw = 0;
  if (pValid) {
    fd = pFill / HW;
    int r = pFill - fd * HW;
    fh = r / W;
    fw = r - fh * W;
  }
  const int icMask = ICp - 1;

  const int mRow = ocBase + wv * 16 + (lane & 15);
  const _Float16* wrow = w + (size_t)mRow * Kpad + ((lane < 16) ? 0 : 8);
  const int koffB = (lane < 16) ? 0 : 8;
  const int nB = lane & 15;

  v8f c0 = {}; v8f c1 = {};

  for (int k0 = 0; k0 < Kpad; k0 += 32) {
    __syncthreads();
    if constexpr (VEC) {
      const int kk0 = k0 + kCol;
      const int tap = kk0 >> icShift;          // constant-divisor-free
      const int ic0 = kk0 & icMask;
      const int kd = tap / 9;
      const int r  = tap - kd * 9;
      const int kh = r / 3;
      const int kw = r - kh * 3;
      const int id = fd + kd - 1, ih = fh + kh - 1, iw = fw + kw - 1;
      const bool ok = pValid && (unsigned)id < (unsigned)D &&
                      (unsigned)ih < (unsigned)H && (unsigned)iw < (unsigned)W;
      if constexpr (ELEM == 8) {
        v8h v = {};
        if (ok) v = *(const v8h*)&x[(size_t)((id * H + ih) * W + iw) * ICp + ic0];
        *(v8h*)&Bs[nFill][kCol] = v;
      } else {
        v4h v = {};
        if (ok) v = *(const v4h*)&x[(size_t)((id * H + ih) * W + iw) * ICp + ic0];
        *(v4h*)&Bs[nFill][kCol] = v;
      }
    } else {
#pragma unroll
      for (int i = 0; i < ELEM; ++i) {
        const int kk = k0 + kCol + i;
        const int tap = kk >> icShift;
        const int ic  = kk & icMask;
        float v = 0.f;
        if (pValid && tap < 27 && ic < ICreal) {
          const int kd = tap / 9;
          const int r  = tap - kd * 9;
          const int kh = r / 3;
          const int kw = r - kh * 3;
          const int id = fd + kd - 1, ih = fh + kh - 1, iw = fw + kw - 1;
          if ((unsigned)id < (unsigned)D && (unsigned)ih < (unsigned)H &&
              (unsigned)iw < (unsigned)W)
            v = (float)x[(size_t)((id * H + ih) * W + iw) * ICp + ic];
        }
        Bs[nFill][kCol + i] = (_Float16)v;
      }
    }
    __syncthreads();

    if (k0 + 32 < Kpad) __builtin_prefetch(wrow + k0 + 32, 0, 0);
    v16h a  = load_fragA(wrow + k0);
    v16h b0 = load_fragB(&Bs[nB][koffB]);
    v16h b1 = load_fragB(&Bs[nB + 16][koffB]);
    c0 = __builtin_amdgcn_wmma_f32_16x16x32_f16(false, a, false, b0,
                                                (short)0, c0, false, false);
    c1 = __builtin_amdgcn_wmma_f32_16x16x32_f16(false, a, false, b1,
                                                (short)0, c1, false, false);
  }

  // Epilogue: per lane the 8 rows are contiguous output channels -> v8h store.
  const int mb = ocBase + wv * 16 + ((lane < 16) ? 0 : 8);
  {
    const int p = p0 + nB;
    if (p < P) {
      v8h o;
#pragma unroll
      for (int j = 0; j < 8; ++j) {
        float v = scale * (c0[j] + bias[mb + j]);
        o[j] = (_Float16)(v > 0.f ? v : 0.f);
      }
      *(v8h*)&y[(size_t)p * OC + mb] = o;
    }
  }
  {
    const int p = p0 + nB + 16;
    if (p < P) {
      v8h o;
#pragma unroll
      for (int j = 0; j < 8; ++j) {
        float v = scale * (c1[j] + bias[mb + j]);
        o[j] = (_Float16)(v > 0.f ? v : 0.f);
      }
      *(v8h*)&y[(size_t)p * OC + mb] = o;
    }
  }
}

// ---------------------------------------------------------------------------
// FC via WMMA: out[n][m] = relu(scale*(sum_k x[n][k]*w[m][k] + bias[m]))
__global__ void __launch_bounds__(128)
fc_wmma(const _Float16* __restrict__ x, const _Float16* __restrict__ w,
        const float* __restrict__ bias, _Float16* __restrict__ y16,
        float* __restrict__ y32, int Nb, int K, int M, float scale) {
  const int lane = threadIdx.x & 31;
  const int wv   = threadIdx.x >> 5;
  const int m0 = blockIdx.x * 64 + wv * 16;
  const int n0 = blockIdx.y * 16;

  const int mRow = m0 + (lane & 15);
  const _Float16* wrow = w + (size_t)mRow * K + ((lane < 16) ? 0 : 8);
  const int nCol = n0 + (lane & 15);
  const bool nOK = (nCol < Nb);
  const _Float16* xrow =
      x + (size_t)(nOK ? nCol : 0) * K + ((lane < 16) ? 0 : 8);

  v8f c = {};
  for (int k0 = 0; k0 < K; k0 += 32) {
    if (k0 + 32 < K) __builtin_prefetch(wrow + k0 + 32, 0, 0);
    v16h a = load_fragA(wrow + k0);
    v16h b = load_fragB(xrow + k0);
    c = __builtin_amdgcn_wmma_f32_16x16x32_f16(false, a, false, b,
                                               (short)0, c, false, false);
  }
  if (nOK) {
    const int mb = m0 + ((lane < 16) ? 0 : 8);
    if (y16) {
      v8h o;
#pragma unroll
      for (int j = 0; j < 8; ++j) {
        float v = scale * (c[j] + bias[mb + j]);
        o[j] = (_Float16)(v > 0.f ? v : 0.f);
      }
      *(v8h*)&y16[(size_t)nCol * M + mb] = o;
    } else {
      v4f o0, o1;
#pragma unroll
      for (int j = 0; j < 4; ++j) {
        float v = scale * (c[j] + bias[mb + j]);
        o0[j] = v > 0.f ? v : 0.f;
        v = scale * (c[j + 4] + bias[mb + j + 4]);
        o1[j] = v > 0.f ? v : 0.f;
      }
      *(v4f*)&y32[(size_t)nCol * M + mb] = o0;
      *(v4f*)&y32[(size_t)nCol * M + mb + 4] = o1;
    }
  }
}

// ---------------------------------------------------------------------------
// Max-pool, NDHWC (c innermost -> coalesced).
__global__ void maxpool3d(const _Float16* __restrict__ x,
                          _Float16* __restrict__ y, int C, int D, int H, int W,
                          int OD, int OH, int OW, int kd, int kh, int kw,
                          int sd, int sh, int sw, int pd, int ph, int pw) {
  const int idx = blockIdx.x * blockDim.x + threadIdx.x;
  const int total = C * OD * OH * OW;
  if (idx >= total) return;
  const int c = idx % C;
  int t = idx / C;
  const int ow = t % OW; t /= OW;
  const int oh = t % OH;
  const int od = t / OH;
  float m = -INFINITY;
  for (int a = 0; a < kd; ++a) {
    const int id = od * sd - pd + a;
    if ((unsigned)id >= (unsigned)D) continue;
    for (int b = 0; b < kh; ++b) {
      const int ih = oh * sh - ph + b;
      if ((unsigned)ih >= (unsigned)H) continue;
      for (int e = 0; e < kw; ++e) {
        const int iw = ow * sw - pw + e;
        if ((unsigned)iw >= (unsigned)W) continue;
        const float v = (float)x[(size_t)((id * H + ih) * W + iw) * C + c];
        m = v > m ? v : m;
      }
    }
  }
  y[idx] = (_Float16)m;
}

// ---------------------------------------------------------------------------
// Conv weight repack: OIDHW f32 -> [OC][Kpad] f16 with K = tap*ICp + ic.
__global__ void cvt_conv_w(const float* __restrict__ src,
                           _Float16* __restrict__ dst, int IC, int ICp,
                           int Kpad, int total) {
  const int i = blockIdx.x * blockDim.x + threadIdx.x;
  if (i >= total) return;
  const int k = i % Kpad;
  const int oc = i / Kpad;
  const int tap = k / ICp;
  const int ic  = k - tap * ICp;
  _Float16 v = (_Float16)0.f;
  if (tap < 27 && ic < IC)
    v = (_Float16)src[((size_t)oc * IC + ic) * 27 + tap];
  dst[i] = v;
}

// fc6 weight repack: K reorder c*16+hw -> hw*512+c (matches NDHWC pool5 flat)
__global__ void cvt_fc6w(const float* __restrict__ src,
                         _Float16* __restrict__ dst, int total) {
  const int i = blockIdx.x * blockDim.x + threadIdx.x;
  if (i >= total) return;
  const int k = i & 8191;
  const int m = i >> 13;
  const int c = k & 511;
  const int hw = k >> 9;
  dst[i] = (_Float16)src[((size_t)m << 13) + c * 16 + hw];
}

// plain f32 -> f16 (fc7)
__global__ void cvt_w(const float* __restrict__ src, _Float16* __restrict__ dst,
                      int total) {
  const int i = blockIdx.x * blockDim.x + threadIdx.x;
  if (i < total) dst[i] = (_Float16)src[i];
}

// (N,seg,t,c,h,w) f32 -> clip-major [t][h][w][4] f16 (ch 3 zero-padded)
__global__ void pack_in(const float* __restrict__ sup,
                        const float* __restrict__ qry,
                        _Float16* __restrict__ dst) {
  const size_t HW = 112 * 112;
  const size_t perClip = (size_t)16 * HW * 4;
  const size_t i = (size_t)blockIdx.x * blockDim.x + threadIdx.x;
  if (i >= 24 * perClip) return;
  const size_t clip = i / perClip;
  size_t r = i - clip * perClip;          // (t*HW + hw)*4 + ch
  const size_t ch = r & 3;
  r >>= 2;
  const size_t t = r / HW;
  const size_t hw = r - t * HW;
  float v = 0.f;
  if (ch < 3) {
    const float* src = (clip < 12) ? sup : qry;
    const size_t vclip = (clip < 12) ? clip : clip - 12;
    v = src[((vclip * 16 + t) * 3 + ch) * HW + hw];
  }
  dst[i] = (_Float16)v;
}

// ---------------------------------------------------------------------------
// Cosine cost + Sinkhorn (4x4, 100 iters) per (query,support) pair.
__global__ void __launch_bounds__(256)
ot_kernel(const float* __restrict__ feat, float* __restrict__ out) {
  const int q = blockIdx.x, s = blockIdx.y;
  const int Dm = 4096;
  __shared__ float red[256];
  __shared__ float dots[16], qn[4], sn[4];
  const int tid = threadIdx.x;

  for (int task = 0; task < 24; ++task) {
    float acc = 0.f;
    if (task < 4) {
      const float* a = feat + (size_t)(12 + q * 4 + task) * Dm;
      for (int k = tid; k < Dm; k += 256) { float v = a[k]; acc += v * v; }
    } else if (task < 8) {
      const float* a = feat + (size_t)(s * 4 + (task - 4)) * Dm;
      for (int k = tid; k < Dm; k += 256) { float v = a[k]; acc += v * v; }
    } else {
      const int i = (task - 8) >> 2, j = (task - 8) & 3;
      const float* a = feat + (size_t)(12 + q * 4 + i) * Dm;
      const float* b = feat + (size_t)(s * 4 + j) * Dm;
      for (int k = tid; k < Dm; k += 256) acc += a[k] * b[k];
    }
    red[tid] = acc;
    __syncthreads();
    for (int off = 128; off > 0; off >>= 1) {
      if (tid < off) red[tid] += red[tid + off];
      __syncthreads();
    }
    if (tid == 0) {
      const float v = red[0];
      if (task < 4) qn[task] = sqrtf(v);
      else if (task < 8) sn[task - 4] = sqrtf(v);
      else dots[task - 8] = v;
    }
    __syncthreads();
  }

  if (tid == 0) {
    float sem[4][4], Km[4][4];
    for (int i = 0; i < 4; ++i)
      for (int j = 0; j < 4; ++j) {
        const float cs = dots[i * 4 + j] / ((qn[i] + 1e-8f) * (sn[j] + 1e-8f));
        sem[i][j] = 1.f - cs;
        const float dt = (float)(i - j) * 0.25f;
        const float pc = expf(-1.f / (dt * dt + 1.f));
        Km[i][j] = expf(-7.f * (sem[i][j] + 0.4f * pc));
      }
    float u[4] = {0.25f, 0.25f, 0.25f, 0.25f}, v[4];
    for (int it = 0; it < 100; ++it) {
      for (int j = 0; j < 4; ++j) {
        float t = 0.f;
        for (int i = 0; i < 4; ++i) t += Km[i][j] * u[i];
        v[j] = 0.25f / (t + 1e-9f);
      }
      for (int i = 0; i < 4; ++i) {
        float t = 0.f;
        for (int j = 0; j < 4; ++j) t += Km[i][j] * v[j];
        u[i] = 0.25f / (t + 1e-9f);
      }
    }
    for (int j = 0; j < 4; ++j) {
      float t = 0.f;
      for (int i = 0; i < 4; ++i) t += Km[i][j] * u[i];
      v[j] = 0.25f / (t + 1e-9f);
    }
    float tc = 0.f;
    for (int i = 0; i < 4; ++i)
      for (int j = 0; j < 4; ++j) tc += u[i] * Km[i][j] * v[j] * sem[i][j];
    out[q * 3 + s] = -tc;
  }
}

// ---------------------------------------------------------------------------
extern "C" void kernel_launch(void* const* d_in, const int* in_sizes, int n_in,
                              void* d_out, int out_size, void* d_ws,
                              size_t ws_size, hipStream_t stream) {
  (void)in_sizes; (void)n_in; (void)out_size; (void)ws_size;
  const float* sup = (const float*)d_in[0];
  const float* qry = (const float*)d_in[1];
  const float BN = 1.0f / sqrtf(1.0f + 1e-5f);

  char* ws = (char*)d_ws;
  size_t off = 0;
  auto alloc = [&](size_t bytes) -> void* {
    void* p = ws + off;
    off = (off + bytes + 255) & ~(size_t)255;
    return p;
  };

  _Float16* wc1  = (_Float16*)alloc((size_t)64 * 128 * 2);
  _Float16* wc2  = (_Float16*)alloc((size_t)128 * 1728 * 2);
  _Float16* wc3a = (_Float16*)alloc((size_t)256 * 3456 * 2);
  _Float16* wc3b = (_Float16*)alloc((size_t)256 * 6912 * 2);
  _Float16* wc4a = (_Float16*)alloc((size_t)512 * 6912 * 2);
  _Float16* wc4b = (_Float16*)alloc((size_t)512 * 13824 * 2);
  _Float16* wc5a = (_Float16*)alloc((size_t)512 * 13824 * 2);
  _Float16* wc5b = (_Float16*)alloc((size_t)512 * 13824 * 2);
  _Float16* wfc6 = (_Float16*)alloc((size_t)4096 * 8192 * 2);
  _Float16* wfc7 = (_Float16*)alloc((size_t)4096 * 4096 * 2);
  _Float16* in16 = (_Float16*)alloc((size_t)24 * 16 * 112 * 112 * 4 * 2);
  _Float16* bufA = (_Float16*)alloc((size_t)16 * 112 * 112 * 64 * 2);
  _Float16* bufB = (_Float16*)alloc((size_t)16 * 112 * 112 * 64 * 2);
  _Float16* featbuf = (_Float16*)alloc((size_t)24 * 8192 * 2);
  _Float16* fc6o    = (_Float16*)alloc((size_t)24 * 4096 * 2);
  float*    featf   = (float*)alloc((size_t)24 * 4096 * 4);

  auto cvtc = [&](const void* src, _Float16* dst, int OC, int IC, int ICp,
                  int Kpad) {
    const int total = OC * Kpad;
    cvt_conv_w<<<(total + 255) / 256, 256, 0, stream>>>((const float*)src, dst,
                                                        IC, ICp, Kpad, total);
  };
  cvtc(d_in[2],  wc1,  64,  3,   4,   128);
  cvtc(d_in[4],  wc2,  128, 64,  64,  1728);
  cvtc(d_in[6],  wc3a, 256, 128, 128, 3456);
  cvtc(d_in[8],  wc3b, 256, 256, 256, 6912);
  cvtc(d_in[10], wc4a, 512, 256, 256, 6912);
  cvtc(d_in[12], wc4b, 512, 512, 512, 13824);
  cvtc(d_in[14], wc5a, 512, 512, 512, 13824);
  cvtc(d_in[16], wc5b, 512, 512, 512, 13824);
  {
    const int t6 = 4096 * 8192;
    cvt_fc6w<<<(t6 + 255) / 256, 256, 0, stream>>>((const float*)d_in[18],
                                                   wfc6, t6);
    const int t7 = 4096 * 4096;
    cvt_w<<<(t7 + 255) / 256, 256, 0, stream>>>((const float*)d_in[20], wfc7,
                                                t7);
  }
  {
    const size_t total = (size_t)24 * 16 * 112 * 112 * 4;
    pack_in<<<(unsigned)((total + 255) / 256), 256, 0, stream>>>(sup, qry,
                                                                 in16);
  }

  // conv launchers: 4-wave scalar-gather (conv1) / 8-wave vector-gather (rest)
  auto conv1L = [&](const _Float16* xin, _Float16* yout, const _Float16* wgt,
                    const void* bias, float scale) {
    const int P = 16 * 112 * 112;
    dim3 g((P + 31) / 32, 64 / 64);
    conv3d_wmma<4, false><<<g, 128, 0, stream>>>(
        xin, wgt, (const float*)bias, yout, 4, 2, 3, 16, 112, 112, 128, 64,
        scale);
  };
  auto convL = [&](const _Float16* xin, _Float16* yout, const _Float16* wgt,
                   const void* bias, int ICp, int icShift, int OC, int D,
                   int H, int W, int Kpad, float scale) {
    const int P = D * H * W;
    dim3 g((P + 31) / 32, OC / 128);
    conv3d_wmma<8, true><<<g, 256, 0, stream>>>(
        xin, wgt, (const float*)bias, yout, ICp, icShift, ICp, D, H, W, Kpad,
        OC, scale);
  };
  auto pool = [&](const _Float16* xin, _Float16* yout, int C, int D, int H,
                  int W, int OD, int OH, int OW, int kd, int kh, int kw,
                  int sd, int sh, int sw, int pd, int ph, int pw) {
    const int total = C * OD * OH * OW;
    maxpool3d<<<(total + 255) / 256, 256, 0, stream>>>(
        xin, yout, C, D, H, W, OD, OH, OW, kd, kh, kw, sd, sh, sw, pd, ph, pw);
  };

  for (int clip = 0; clip < 24; ++clip) {
    const _Float16* x0 = in16 + (size_t)clip * 16 * 112 * 112 * 4;
    conv1L(x0, bufA, wc1, d_in[3], BN);
    pool(bufA, bufB, 64, 16, 112, 112, 16, 56, 56, 1, 2, 2, 1, 2, 2, 0, 0, 0);
    convL(bufB, bufA, wc2, d_in[5], 64, 6, 128, 16, 56, 56, 1728, BN);
    pool(bufA, bufB, 128, 16, 56, 56, 8, 28, 28, 2, 2, 2, 2, 2, 2, 0, 0, 0);
    convL(bufB, bufA, wc3a, d_in[7], 128, 7, 256, 8, 28, 28, 3456, 1.0f);
    convL(bufA, bufB, wc3b, d_in[9], 256, 8, 256, 8, 28, 28, 6912, BN);
    pool(bufB, bufA, 256, 8, 28, 28, 4, 14, 14, 2, 2, 2, 2, 2, 2, 0, 0, 0);
    convL(bufA, bufB, wc4a, d_in[11], 256, 8, 512, 4, 14, 14, 6912, 1.0f);
    convL(bufB, bufA, wc4b, d_in[13], 512, 9, 512, 4, 14, 14, 13824, BN);
    pool(bufA, bufB, 512, 4, 14, 14, 2, 7, 7, 2, 2, 2, 2, 2, 2, 0, 0, 0);
    convL(bufB, bufA, wc5a, d_in[15], 512, 9, 512, 2, 7, 7, 13824, 1.0f);
    convL(bufA, bufB, wc5b, d_in[17], 512, 9, 512, 2, 7, 7, 13824, BN);
    // pool5 (2,2,2)/(2,2,2) pad(0,1,1): [2][7][7][512] -> [1][4][4][512]
    pool(bufB, featbuf + (size_t)clip * 8192, 512, 2, 7, 7, 1, 4, 4, 2, 2, 2,
         2, 2, 2, 0, 1, 1);
  }

  // fc6: [24,8192]->[24,4096] f16 ; fc7: ->[24,4096] f32
  fc_wmma<<<dim3(4096 / 64, 2), 128, 0, stream>>>(
      featbuf, wfc6, (const float*)d_in[19], fc6o, nullptr, 24, 8192, 4096, BN);
  fc_wmma<<<dim3(4096 / 64, 2), 128, 0, stream>>>(
      fc6o, wfc7, (const float*)d_in[21], nullptr, featf, 24, 4096, 4096, BN);

  ot_kernel<<<dim3(3, 3), 256, 0, stream>>>(featf, (float*)d_out);
}
